// CPPN_8022998909389
// MI455X (gfx1250) — matrix operand order repaired and checked by reading, not compiled, not run on hardware
//
#include <hip/hip_runtime.h>

typedef __attribute__((ext_vector_type(16))) _Float16 v16h;
typedef __attribute__((ext_vector_type(2)))  __fp16   v2fp16;
typedef __attribute__((ext_vector_type(8)))  float    v8f;
typedef __attribute__((ext_vector_type(2)))  float    v2f;

#define BN_TOTAL (4 * 262144)   // B*N rows
#define HID 64
#define NOUT 3

#if defined(__has_builtin)
#if __has_builtin(__builtin_amdgcn_wmma_f32_16x16x4_f32)
#define HAVE_WMMA_F32X4 1
#endif
#if __has_builtin(__builtin_amdgcn_tanhf)
#define HAVE_TANH 1
#endif
#endif

// ---------- helpers ----------
__device__ __forceinline__ unsigned encf(float f) {
  unsigned u = __float_as_uint(f);
  return (u & 0x80000000u) ? ~u : (u | 0x80000000u);   // order-preserving float->uint
}
__device__ __forceinline__ float decf(unsigned u) {
  unsigned b = (u & 0x80000000u) ? (u & 0x7FFFFFFFu) : ~u;
  return __uint_as_float(b);
}
__device__ __forceinline__ float tanh_fast(float x) {
#ifdef HAVE_TANH
  return __builtin_amdgcn_tanhf(x);                    // v_tanh_f32 (CDNA5 TRANS op)
#else
  float ax = __builtin_fabsf(x);
  float e  = __expf(-2.0f * ax);                       // v_exp_f32, no overflow
  float t  = (1.0f - e) * __builtin_amdgcn_rcpf(1.0f + e);  // v_rcp_f32, no IEEE div seq
  return __builtin_copysignf(t, x);
#endif
}
__device__ __forceinline__ unsigned pk2(float a, float b) {
  union { v2fp16 h; unsigned u; } c;
  c.h = __builtin_amdgcn_cvt_pkrtz(a, b);        // (a -> lo half, b -> hi half)
  return c.u;
}

// Build a 32xK-chunk f16 B operand (K on VGPRs/halves, pixel on lane) from two
// f32 C tiles' packed dwords. p0 = m-tile supplying K=0..15, p1 = K=16..31.
union BFU { v16h h; unsigned u[8]; };
__device__ __forceinline__ v16h buildB(const unsigned (&p0)[4], const unsigned (&p1)[4], int lane) {
  BFU bf;
  const bool lo = lane < 16;
#pragma unroll
  for (int j = 0; j < 4; ++j) {
    unsigned s0 = (unsigned)__shfl_xor((int)p0[j], 16, 32);
    unsigned s1 = (unsigned)__shfl_xor((int)p1[j], 16, 32);
    bf.u[j]     = lo ? p0[j] : s1;   // VGPR j   : K = 2j,2j+1   (m 0..7  of the chunk)
    bf.u[4 + j] = lo ? s0    : p1[j];// VGPR j+4 : K = 8+2j,...  (m 8..15 of the chunk)
  }
  return bf.h;
}

// ---------- kernel 0: weight fragment prep + min/max init ----------
// frag layout: op in [0,10): [lane][16 halves].  ops 0..7 = W2^T (t=op>>1 m-tile, c=op&1 K-chunk),
// ops 8..9 = W3^T padded to M=16 (c = op-8).
__global__ void cppn_setup_kernel(const float* __restrict__ W2, const float* __restrict__ W3,
                                  unsigned* __restrict__ mm, _Float16* __restrict__ frag) {
  const int blk = blockIdx.x;
  const int L   = threadIdx.x;      // 32 threads
  if (blk == 10) {
    if (L == 0) { mm[0] = 0xFFFFFFFFu; mm[1] = 0u; }   // min-enc neutral, max-enc neutral
    return;
  }
  const int mrow = L & 15;
  const int hi   = L >> 4;
  _Float16* dst = frag + (size_t)blk * 512 + (size_t)L * 16;
  if (blk < 8) {
    const int t = blk >> 1, c = blk & 1;
    const int m = t * 16 + mrow;
#pragma unroll
    for (int h = 0; h < 16; ++h) {
      // A f16 16x32, contiguous k order used here (same as buildB's B): halves 0-15 -> k = 16*hi + 0..15
      const int k = c * 32 + 16 * hi + h;
      dst[h] = (_Float16)W2[k * HID + m];               // A[m][k] = W2[k][m]
    }
  } else {
    const int c = blk - 8;
#pragma unroll
    for (int h = 0; h < 16; ++h) {
      const int k = c * 32 + 16 * hi + h;
      dst[h] = (mrow < NOUT) ? (_Float16)W3[k * NOUT + mrow] : (_Float16)0.0f;
    }
  }
}

// ---------- kernel 1: fused 3-layer MLP, one wave = 16 pixels ----------
__global__ __launch_bounds__(256) void cppn_mlp_kernel(
    const float* __restrict__ x,  const float* __restrict__ W1,
    const float* __restrict__ b1, const float* __restrict__ b2,
    const float* __restrict__ b3, const _Float16* __restrict__ frag,
    float* __restrict__ out, unsigned* __restrict__ mm)
{
  const int lane = threadIdx.x & 31;
  const int wave = threadIdx.x >> 5;
  const int tile = blockIdx.x * 8 + wave;
  const int row0 = tile * 16;
  const int px   = lane & 15;          // pixel (N) on lanes
  const int hi   = lane >> 4;

  // ---- layer 1: h1 = tanh(W1^T(64x4) @ x^T(4x16) + b1), 4 M-tiles ----
  v8f h1[4];
#ifdef HAVE_WMMA_F32X4
  v2f bx;
  { const float* xp = x + (size_t)(row0 + px) * 4 + 2 * hi;   // B f32 4x16: k = 2*hi+{0,1}
    bx.x = xp[0]; bx.y = xp[1]; }
#else
  float xr[4];
  { const float* xp = x + (size_t)(row0 + px) * 4;
#pragma unroll
    for (int k = 0; k < 4; ++k) xr[k] = xp[k]; }
#endif
#pragma unroll
  for (int t = 0; t < 4; ++t) {
    v8f c;
    const float* bp = b1 + t * 16 + 8 * hi;      // C layout: m = v + 8*hi
#pragma unroll
    for (int v = 0; v < 8; ++v) c[v] = bp[v];
#ifdef HAVE_WMMA_F32X4
    v2f a;                                       // A f32 16x4: m = lane%16, k = 2*hi+{0,1}
    { const int m = t * 16 + px;
      a.x = W1[(2 * hi + 0) * HID + m];
      a.y = W1[(2 * hi + 1) * HID + m]; }
    c = __builtin_amdgcn_wmma_f32_16x16x4_f32(false, a, false, bx, (short)0, c, false, false);
#else
#pragma unroll
    for (int v = 0; v < 8; ++v) {
      const int m = t * 16 + v + 8 * hi;
      float acc = c[v];
#pragma unroll
      for (int k = 0; k < 4; ++k) acc = __builtin_fmaf(W1[k * HID + m], xr[k], acc);
      c[v] = acc;
    }
#endif
#pragma unroll
    for (int v = 0; v < 8; ++v) h1[t][v] = tanh_fast(c[v]);
  }

  // ---- layer 2: h2 = tanh(W2^T(64x64) @ h1 + b2) via f16 WMMA ----
  unsigned p1[4][4];
#pragma unroll
  for (int t = 0; t < 4; ++t)
#pragma unroll
    for (int j = 0; j < 4; ++j) p1[t][j] = pk2(h1[t][2 * j], h1[t][2 * j + 1]);
  v16h Bf[2];
  Bf[0] = buildB(p1[0], p1[1], lane);            // K = 0..31
  Bf[1] = buildB(p1[2], p1[3], lane);            // K = 32..63

  v8f h2[4];
#pragma unroll
  for (int t = 0; t < 4; ++t) {
    v8f c;
    const float* bp = b2 + t * 16 + 8 * hi;
#pragma unroll
    for (int v = 0; v < 8; ++v) c[v] = bp[v];
#pragma unroll
    for (int cc = 0; cc < 2; ++cc) {
      const v16h A = *(const v16h*)(frag + (size_t)(t * 2 + cc) * 512 + (size_t)lane * 16);
      c = __builtin_amdgcn_wmma_f32_16x16x32_f16(false, A, false, Bf[cc], (short)0, c, false, false);
    }
#pragma unroll
    for (int v = 0; v < 8; ++v) h2[t][v] = tanh_fast(c[v]);
  }

  // ---- layer 3: out = W3^T(3x64 padded to 16) @ h2 + b3 ----
  unsigned p2[4][4];
#pragma unroll
  for (int t = 0; t < 4; ++t)
#pragma unroll
    for (int j = 0; j < 4; ++j) p2[t][j] = pk2(h2[t][2 * j], h2[t][2 * j + 1]);
  v16h B3[2];
  B3[0] = buildB(p2[0], p2[1], lane);
  B3[1] = buildB(p2[2], p2[3], lane);

  v8f o;
#pragma unroll
  for (int v = 0; v < 8; ++v) o[v] = (v < NOUT && hi == 0) ? b3[v] : 0.0f;
#pragma unroll
  for (int cc = 0; cc < 2; ++cc) {
    const v16h A = *(const v16h*)(frag + (size_t)(8 + cc) * 512 + (size_t)lane * 16);
    o = __builtin_amdgcn_wmma_f32_16x16x32_f16(false, A, false, B3[cc], (short)0, o, false, false);
  }

  // ---- store raw outputs [NO][B*N] + wave min/max -> global atomics ----
  float vmin = __builtin_inff();
  float vmax = -__builtin_inff();
  if (hi == 0) {
#pragma unroll
    for (int v = 0; v < NOUT; ++v) {             // lane<16: m = v = output channel
      const float val = o[v];
      out[(size_t)v * BN_TOTAL + row0 + px] = val;
      vmin = __builtin_fminf(vmin, val);
      vmax = __builtin_fmaxf(vmax, val);
    }
  }
#pragma unroll
  for (int off = 16; off >= 1; off >>= 1) {
    vmin = __builtin_fminf(vmin, __shfl_xor(vmin, off, 32));
    vmax = __builtin_fmaxf(vmax, __shfl_xor(vmax, off, 32));
  }
  if (lane == 0) {
    atomicMin(&mm[0], encf(vmin));
    atomicMax(&mm[1], encf(vmax));
  }
}

// ---------- kernel 2: in-place global min-max normalize + clip ----------
__global__ __launch_bounds__(256) void cppn_norm_kernel(float* __restrict__ out,
                                                        const unsigned* __restrict__ mm) {
  const int i = blockIdx.x * blockDim.x + threadIdx.x;   // over float4, exact coverage
  const float mn  = decf(mm[0]);
  const float mx  = decf(mm[1]);
  const float inv = __builtin_amdgcn_rcpf(__builtin_fmaxf(mx - mn, 1e-30f));
  float4* p = (float4*)out;
  float4 v = p[i];
  v.x = __builtin_fminf(__builtin_fmaxf((v.x - mn) * inv, 0.0f), 1.0f);
  v.y = __builtin_fminf(__builtin_fmaxf((v.y - mn) * inv, 0.0f), 1.0f);
  v.z = __builtin_fminf(__builtin_fmaxf((v.z - mn) * inv, 0.0f), 1.0f);
  v.w = __builtin_fminf(__builtin_fmaxf((v.w - mn) * inv, 0.0f), 1.0f);
  p[i] = v;
}

// ---------- launch ----------
extern "C" void kernel_launch(void* const* d_in, const int* in_sizes, int n_in,
                              void* d_out, int out_size, void* d_ws, size_t ws_size,
                              hipStream_t stream) {
  const float* x  = (const float*)d_in[0];
  const float* W1 = (const float*)d_in[1];
  const float* b1 = (const float*)d_in[2];
  const float* W2 = (const float*)d_in[3];
  const float* b2 = (const float*)d_in[4];
  const float* W3 = (const float*)d_in[5];
  const float* b3 = (const float*)d_in[6];
  (void)in_sizes; (void)n_in; (void)out_size; (void)ws_size;

  unsigned*  mm   = (unsigned*)d_ws;                     // 2 words: enc(min), enc(max)
  _Float16*  frag = (_Float16*)((char*)d_ws + 256);      // 10 ops * 32 lanes * 32B = 10 KB

  cppn_setup_kernel<<<11, 32, 0, stream>>>(W2, W3, mm, frag);
  cppn_mlp_kernel<<<BN_TOTAL / 16 / 8, 256, 0, stream>>>(x, W1, b1, b2, b3,
                                                         (const _Float16*)frag,
                                                         (float*)d_out, mm);
  cppn_norm_kernel<<<(3 * BN_TOTAL / 4) / 256, 256, 0, stream>>>((float*)d_out, mm);
}
